// MambaResidualBlock_37065567764782
// MI455X (gfx1250) — matrix operand order repaired and checked by reading, not compiled, not run on hardware
//
#include <hip/hip_runtime.h>

// ---------------------------------------------------------------------------
// Mamba residual block for MI455X (gfx1250), wave32.
// bf16 WMMA GEMMs (f32 accumulate), fp32 elementwise + scan recurrence.
// TDM (tensor_load_to_lds) stages the per-timestep B/C slab for the scan.
// B=2, L=2048, D_MODEL=1024, D_INNER=2048, D_STATE=16, DT_RANK=64, D_CONV=4
// ---------------------------------------------------------------------------

#define B_      2
#define L_      2048
#define DMODEL  1024
#define DINNER  2048
#define DSTATE  16
#define DTRANK  64
#define DCONV   4
#define M_      (B_ * L_)         // 4096 token rows
#define XCOLS   (DTRANK + 2 * DSTATE)   // 96
#define CHUNK   64                // scan timesteps staged per TDM op

typedef __attribute__((ext_vector_type(8)))  __bf16   v8bf;
typedef __attribute__((ext_vector_type(16))) __bf16   v16bf;
typedef __attribute__((ext_vector_type(8)))  float    v8f;
typedef __attribute__((ext_vector_type(4)))  unsigned v4u;
typedef __attribute__((ext_vector_type(4)))  int      v4i;
typedef __attribute__((ext_vector_type(8)))  int      v8i;

union ABfrag { v16bf v; v8bf h[2]; };

static __device__ __forceinline__ __bf16 f2bf(float f) {
    unsigned u = __builtin_bit_cast(unsigned, f);
    unsigned r = u + 0x7FFFu + ((u >> 16) & 1u);   // round-to-nearest-even
    unsigned short hv = (unsigned short)(r >> 16);
    return __builtin_bit_cast(__bf16, hv);
}

static __device__ __forceinline__ float silu(float x) {
    return x / (1.0f + __expf(-x));
}

// --------------------------- elementwise helpers ---------------------------

__global__ void cvt_f32_bf16_kernel(const float* __restrict__ s,
                                    __bf16* __restrict__ d, int n) {
    int i = blockIdx.x * blockDim.x + threadIdx.x;
    if (i < n) d[i] = f2bf(s[i]);
}

// RMSNorm over DMODEL per token; emit bf16 normalized activations.
__global__ __launch_bounds__(256)
void rmsnorm_bf16_kernel(const float* __restrict__ x,
                         const float* __restrict__ w,
                         __bf16* __restrict__ out) {
    int row = blockIdx.x;                       // 0..M_-1
    const float* xr = x + (size_t)row * DMODEL;
    float s = 0.f;
    for (int i = threadIdx.x; i < DMODEL; i += 256) { float v = xr[i]; s += v * v; }
    __shared__ float red[256];
    red[threadIdx.x] = s;
    __syncthreads();
    for (int st = 128; st > 0; st >>= 1) {
        if (threadIdx.x < st) red[threadIdx.x] += red[threadIdx.x + st];
        __syncthreads();
    }
    float scale = rsqrtf(red[0] * (1.0f / DMODEL) + 1e-5f);
    for (int i = threadIdx.x; i < DMODEL; i += 256)
        out[(size_t)row * DMODEL + i] = f2bf(xr[i] * scale * w[i]);
}

// ------------------------------- WMMA GEMM ---------------------------------
// C[M x N] (f32) = A[M x K] (bf16 row-major) * W[N x K]^T (bf16 row-major).
// Each wave computes a 32x32 tile as 2x2 WMMA sub-tiles (4 accumulators):
// 8 fragment loads feed 4 v_wmma per K-step. 8 waves/block -> 64x128 block
// tile. Requires M%64==0, N%32==0 (true for all four GEMMs here).
// Fragment layouts per CDNA5 ISA 7.12.2.
__global__ __launch_bounds__(256)
void gemm_bf16_wmma_kernel(const __bf16* __restrict__ A,
                           const __bf16* __restrict__ W,
                           float* __restrict__ C,
                           const float* __restrict__ resid,
                           int M, int N, int K) {
    int wave = threadIdx.x >> 5;
    int lane = threadIdx.x & 31;
    int mbase = blockIdx.y * 64 + (wave >> 2) * 32;
    int nbase = blockIdx.x * 128 + (wave & 3) * 32;
    if (mbase >= M || nbase >= N) return;       // wave-uniform exit

    int lh = lane >> 4;        // lane half selects K sub-range
    int ln = lane & 15;        // row (A) / column (B) index within tile

    const __bf16* a0 = A + (size_t)(mbase + ln) * K;
    const __bf16* a1 = a0 + (size_t)16 * K;
    const __bf16* b0 = W + (size_t)(nbase + ln) * K;   // B column n = W row n
    const __bf16* b1 = b0 + (size_t)16 * K;

    v8f acc00 = {}, acc01 = {}, acc10 = {}, acc11 = {};
    for (int k = 0; k < K; k += 32) {
        if (k + 32 < K) {      // next K-step -> global_prefetch_b8, WGP scope
            __builtin_prefetch(a0 + k + 32, 0, 3);
            __builtin_prefetch(b0 + k + 32, 0, 3);
            __builtin_prefetch(b1 + k + 32, 0, 3);
        }
        ABfrag fa0, fa1, fb0, fb1;
        // A 16x32: lane<16 holds K {0..7, 16..23}; lane>=16 holds {8..15, 24..31}
        fa0.h[0] = *(const v8bf*)(a0 + k + lh * 8);
        fa0.h[1] = *(const v8bf*)(a0 + k + 16 + lh * 8);
        fa1.h[0] = *(const v8bf*)(a1 + k + lh * 8);
        fa1.h[1] = *(const v8bf*)(a1 + k + 16 + lh * 8);
        // B 32x16: lane<16 holds K {0..7, 16..23}; lane>=16 holds {8..15, 24..31}
        fb0.h[0] = *(const v8bf*)(b0 + k + lh * 8);
        fb0.h[1] = *(const v8bf*)(b0 + k + 16 + lh * 8);
        fb1.h[0] = *(const v8bf*)(b1 + k + lh * 8);
        fb1.h[1] = *(const v8bf*)(b1 + k + 16 + lh * 8);
        acc00 = __builtin_amdgcn_wmma_f32_16x16x32_bf16(false, fa0.v, false, fb0.v, (short)0, acc00, false, false);
        acc01 = __builtin_amdgcn_wmma_f32_16x16x32_bf16(false, fa0.v, false, fb1.v, (short)0, acc01, false, false);
        acc10 = __builtin_amdgcn_wmma_f32_16x16x32_bf16(false, fa1.v, false, fb0.v, (short)0, acc10, false, false);
        acc11 = __builtin_amdgcn_wmma_f32_16x16x32_bf16(false, fa1.v, false, fb1.v, (short)0, acc11, false, false);
    }

    // f32 C/D layout: VGPR r -> M = r + 8*(lane>=16), N = lane&15
#pragma unroll
    for (int r = 0; r < 8; ++r) {
        int row0 = mbase + lh * 8 + r;
        int col0 = nbase + ln;
        size_t i00 = (size_t)row0 * N + col0;
        size_t i01 = i00 + 16;
        size_t i10 = i00 + (size_t)16 * N;
        size_t i11 = i10 + 16;
        float v00 = acc00[r], v01 = acc01[r], v10 = acc10[r], v11 = acc11[r];
        if (resid) {
            v00 += resid[i00]; v01 += resid[i01];
            v10 += resid[i10]; v11 += resid[i11];
        }
        C[i00] = v00; C[i01] = v01; C[i10] = v10; C[i11] = v11;
    }
}

// ------------------------- depthwise conv + SiLU ---------------------------
// u lives in xz[:, 0:DINNER] (row stride 2*DINNER). Causal conv over l, tap 4.
__global__ __launch_bounds__(256)
void conv_silu_kernel(const float* __restrict__ xz,
                      const float* __restrict__ Wc,
                      const float* __restrict__ bc,
                      float* __restrict__ uc,
                      __bf16* __restrict__ ucbf) {
    int d = blockIdx.x * 256 + threadIdx.x;     // 0..DINNER-1
    int l = blockIdx.y;
    int b = blockIdx.z;
    float acc = bc[d];
#pragma unroll
    for (int j = 0; j < DCONV; ++j) {
        int ll = l - (DCONV - 1) + j;
        if (ll >= 0)
            acc += Wc[d * DCONV + j] *
                   xz[((size_t)(b * L_ + ll)) * (2 * DINNER) + d];
    }
    float s = silu(acc);
    size_t idx = ((size_t)(b * L_ + l)) * DINNER + d;
    uc[idx] = s;
    ucbf[idx] = f2bf(s);
}

// Slice dt-rank columns of x_dbl [M x 96] into a dense bf16 [M x 64] matrix.
__global__ void dtslice_kernel(const float* __restrict__ xdbl,
                               __bf16* __restrict__ dtbf) {
    int i = blockIdx.x * blockDim.x + threadIdx.x;
    if (i >= M_ * DTRANK) return;
    int m = i / DTRANK, r = i - m * DTRANK;
    dtbf[i] = f2bf(xdbl[(size_t)m * XCOLS + r]);
}

// dt = softplus(dt_lin + b_dt), in place.
__global__ void softplus_kernel(float* __restrict__ dt,
                                const float* __restrict__ b_dt) {
    int i = blockIdx.x * blockDim.x + threadIdx.x;
    if (i >= M_ * DINNER) return;
    int d = i & (DINNER - 1);
    float v = dt[i] + b_dt[d];
    dt[i] = (v > 20.f) ? v : log1pf(__expf(v));
}

// ---------------------------- selective scan -------------------------------
// One thread per (b, d) channel; h[16] in registers; sequential over L.
// Per 64-step chunk, the strided [CHUNK x 32] B/C slab of x_dbl (cols 64..95,
// row stride 96 f32) is compacted into LDS. On toolchains exposing the TDM
// builtin, wave 0 issues a single tensor_load_to_lds per chunk (TENSORcnt);
// otherwise a cooperative load is used. Two barriers per 64 steps.
// y overwrites dt in place (read-before-write within the owning thread).
#if defined(__AMDGCN__) && defined(__has_builtin)
#if __has_builtin(__builtin_amdgcn_tensor_load_to_lds) && \
    __has_builtin(__builtin_amdgcn_s_wait_tensorcnt)
#define USE_TDM 1
#endif
#endif

#if defined(USE_TDM)
#warning "CDNA5 TDM path ENABLED: tensor_load_to_lds in scan_kernel (device pass)"
#else
#warning "CDNA5 TDM path disabled in this pass: cooperative LDS load fallback"
#endif

__global__ __launch_bounds__(256)
void scan_kernel(const float* __restrict__ xdbl,
                 const float* __restrict__ A_log,
                 float* __restrict__ dty,          // in: dt, out: y
                 const float* __restrict__ uc) {
    int d = blockIdx.x * 256 + threadIdx.x;       // 0..DINNER-1
    int b = blockIdx.y;
    float A2[DSTATE], h[DSTATE];
#pragma unroll
    for (int n = 0; n < DSTATE; ++n) {
        A2[n] = -__expf(A_log[(size_t)d * DSTATE + n]) * 1.4426950408889634f;
        h[n] = 0.f;
    }
    // Only __shared__ object in this kernel -> LDS byte offset 0 (used as the
    // D# group0 lds_addr for the TDM path).
    __shared__ float sBC[CHUNK][2 * DSTATE];      // 8 KB

    for (int c = 0; c < L_ / CHUNK; ++c) {
        __syncthreads();                          // previous chunk consumed
#if defined(USE_TDM)
        if ((threadIdx.x >> 5) == 0) {            // wave 0 issues the DMA
            unsigned long long ga = (unsigned long long)(uintptr_t)xdbl +
                ((size_t)b * L_ * XCOLS + (size_t)c * CHUNK * XCOLS + DTRANK) * 4ull;
            v4u g0;
            g0[0] = 1u;                                   // count=1 (valid), normal mode
            g0[1] = 0u;                                   // lds_addr = 0 (sBC)
            g0[2] = (unsigned)(ga & 0xFFFFFFFFu);         // global_addr[31:0]
            g0[3] = (unsigned)((ga >> 32) & 0x1FFFFFFu)   // global_addr[56:32]
                    | (2u << 30);                         // type = 2 ("image")
            v8i g1;
            g1[0] = (2 << 16);                 // data_size = 4B; mask=0; no iter/pad
            g1[1] = (int)((2 * DSTATE) << 16); // tensor_dim0 = 32 (lo 16 bits)
            g1[2] = (CHUNK << 16);             // tensor_dim0 hi=0 | tensor_dim1 = CHUNK
            g1[3] = ((2 * DSTATE) << 16);      // tensor_dim1 hi=0 | tile_dim0 = 32
            g1[4] = CHUNK;                     // tile_dim1 = CHUNK, tile_dim2 = 0
            g1[5] = XCOLS;                     // tensor_dim0_stride = 96 (lo 32)
            g1[6] = 0;                         // stride hi | tensor_dim1_stride lo
            g1[7] = 0;
            v4i gz = {0, 0, 0, 0};             // 2D tensor: groups 2/3 unused
#if __clang_major__ >= 23
            v8i gz8 = {0, 0, 0, 0, 0, 0, 0, 0};
            __builtin_amdgcn_tensor_load_to_lds(g0, g1, gz, gz, gz8, 0);
#else
            __builtin_amdgcn_tensor_load_to_lds(g0, g1, gz, gz, 0);
#endif
            __builtin_amdgcn_s_wait_tensorcnt(0);
        }
#else
        for (int i = threadIdx.x; i < CHUNK * 2 * DSTATE; i += 256) {
            int lrow = i >> 5, col = i & 31;
            sBC[lrow][col] =
                xdbl[((size_t)(b * L_ + c * CHUNK + lrow)) * XCOLS + DTRANK + col];
        }
#endif
        __syncthreads();

        for (int t = 0; t < CHUNK; ++t) {
            size_t m = (size_t)b * L_ + c * CHUNK + t;
            size_t idx = m * DINNER + d;
            float dt = dty[idx];
            float du = dt * uc[idx];
            float y = 0.f;
#pragma unroll
            for (int n = 0; n < DSTATE; ++n) {
                float dA = exp2f(dt * A2[n]);      // exp(dt*A) via v_exp_f32
                h[n] = dA * h[n] + du * sBC[t][n];
                y += h[n] * sBC[t][DSTATE + n];
            }
            dty[idx] = y;
        }
    }
}

// y_gated = (y + u*D) * silu(z);  z = xz[:, DINNER:2*DINNER]. Emit bf16.
__global__ void gate_kernel(const float* __restrict__ y,
                            const float* __restrict__ uc,
                            const float* __restrict__ xz,
                            const float* __restrict__ Dp,
                            __bf16* __restrict__ ybf) {
    int i = blockIdx.x * blockDim.x + threadIdx.x;
    if (i >= M_ * DINNER) return;
    int m = i / DINNER, d = i - m * DINNER;
    float z = xz[(size_t)m * (2 * DINNER) + DINNER + d];
    ybf[i] = f2bf((y[i] + uc[i] * Dp[d]) * silu(z));
}

// ------------------------------- launcher ----------------------------------

extern "C" void kernel_launch(void* const* d_in, const int* in_sizes, int n_in,
                              void* d_out, int out_size, void* d_ws, size_t ws_size,
                              hipStream_t stream) {
    const float* x      = (const float*)d_in[0];
    const float* norm_w = (const float*)d_in[1];
    const float* W_in   = (const float*)d_in[2];
    const float* W_conv = (const float*)d_in[3];
    const float* b_conv = (const float*)d_in[4];
    const float* W_xprj = (const float*)d_in[5];
    const float* W_dt   = (const float*)d_in[6];
    const float* b_dt   = (const float*)d_in[7];
    const float* A_log  = (const float*)d_in[8];
    const float* Dp     = (const float*)d_in[9];
    const float* W_out  = (const float*)d_in[10];
    float* out = (float*)d_out;
    char*  ws  = (char*)d_ws;

    // workspace layout (256B aligned)
    size_t o = 0;
    auto alloc = [&](size_t bytes) { size_t r = o; o += (bytes + 255) & ~(size_t)255; return r; };
    __bf16* win_bf  = (__bf16*)(ws + alloc((size_t)2 * DINNER * DMODEL * 2));   // 8 MB
    __bf16* wxp_bf  = (__bf16*)(ws + alloc((size_t)XCOLS * DINNER * 2));
    __bf16* wdt_bf  = (__bf16*)(ws + alloc((size_t)DINNER * DTRANK * 2));
    __bf16* wout_bf = (__bf16*)(ws + alloc((size_t)DMODEL * DINNER * 2));
    __bf16* xn_bf   = (__bf16*)(ws + alloc((size_t)M_ * DMODEL * 2));
    float*  xz      = (float* )(ws + alloc((size_t)M_ * 2 * DINNER * 4));       // 64 MB
    float*  uc      = (float* )(ws + alloc((size_t)M_ * DINNER * 4));
    __bf16* ucbf    = (__bf16*)(ws + alloc((size_t)M_ * DINNER * 2));
    float*  xdbl    = (float* )(ws + alloc((size_t)M_ * XCOLS * 4));
    __bf16* dtbf    = (__bf16*)(ws + alloc((size_t)M_ * DTRANK * 2));
    float*  dty     = (float* )(ws + alloc((size_t)M_ * DINNER * 4));           // dt then y
    __bf16* ybf     = (__bf16*)(ws + alloc((size_t)M_ * DINNER * 2));
    (void)ws_size; (void)in_sizes; (void)n_in; (void)out_size;

    // 1) weight conversions to bf16
    cvt_f32_bf16_kernel<<<(2 * DINNER * DMODEL) / 256, 256, 0, stream>>>(W_in, win_bf, 2 * DINNER * DMODEL);
    cvt_f32_bf16_kernel<<<(XCOLS * DINNER) / 256, 256, 0, stream>>>(W_xprj, wxp_bf, XCOLS * DINNER);
    cvt_f32_bf16_kernel<<<(DINNER * DTRANK) / 256, 256, 0, stream>>>(W_dt, wdt_bf, DINNER * DTRANK);
    cvt_f32_bf16_kernel<<<(DMODEL * DINNER) / 256, 256, 0, stream>>>(W_out, wout_bf, DMODEL * DINNER);

    // 2) RMSNorm -> bf16
    rmsnorm_bf16_kernel<<<M_, 256, 0, stream>>>(x, norm_w, xn_bf);

    // 3) in_proj: xz[M x 4096] = xn[M x 1024] * W_in^T
    gemm_bf16_wmma_kernel<<<dim3(2 * DINNER / 128, M_ / 64), 256, 0, stream>>>(
        xn_bf, win_bf, xz, nullptr, M_, 2 * DINNER, DMODEL);

    // 4) depthwise causal conv + SiLU on u = xz[:, :DINNER]
    conv_silu_kernel<<<dim3(DINNER / 256, L_, B_), 256, 0, stream>>>(
        xz, W_conv, b_conv, uc, ucbf);

    // 5) x_proj: x_dbl[M x 96] = u_c * W_xproj^T  (96 cols -> 3 wave-columns)
    gemm_bf16_wmma_kernel<<<dim3(1, M_ / 64), 256, 0, stream>>>(
        ucbf, wxp_bf, xdbl, nullptr, M_, XCOLS, DINNER);

    // 6) dt GEMM: dt_lin[M x 2048] = x_dbl[:, :64] * W_dt^T ; then softplus(+b_dt)
    dtslice_kernel<<<(M_ * DTRANK) / 256, 256, 0, stream>>>(xdbl, dtbf);
    gemm_bf16_wmma_kernel<<<dim3(DINNER / 128, M_ / 64), 256, 0, stream>>>(
        dtbf, wdt_bf, dty, nullptr, M_, DINNER, DTRANK);
    softplus_kernel<<<(M_ * DINNER) / 256, 256, 0, stream>>>(dty, b_dt);

    // 7) selective scan (sequential over L); y overwrites dt in place
    scan_kernel<<<dim3(DINNER / 256, B_), 256, 0, stream>>>(xdbl, A_log, dty, uc);

    // 8) gate: (y + u*D) * silu(z) -> bf16
    gate_kernel<<<(M_ * DINNER) / 256, 256, 0, stream>>>(dty, uc, xz, Dp, ybf);

    // 9) out_proj + residual: out = y_g * W_out^T + x
    gemm_bf16_wmma_kernel<<<dim3(DMODEL / 128, M_ / 64), 256, 0, stream>>>(
        ybf, wout_bf, out, x, M_, DMODEL, DINNER);
}
